// EvolvingSystem_82188494177067
// MI455X (gfx1250) — matrix operand order repaired and checked by reading, not compiled, not run on hardware
//
#include <hip/hip_runtime.h>

typedef _Float16 h16;
typedef __attribute__((ext_vector_type(4)))  _Float16 v4h;
typedef __attribute__((ext_vector_type(8)))  _Float16 v8h;
typedef __attribute__((ext_vector_type(16))) _Float16 v16h;
typedef __attribute__((ext_vector_type(8)))  float    v8f;

#define B_    2048
#define KC    64
#define DIM   128
#define R1_   1024
#define OUT_  512
#define NSTEP (KC * (R1_ / 32))   // 2048 fused (k, i-chunk) pipeline steps

// gfx1250 async global->LDS path (ASYNCcnt). The s_wait builtin is confirmed by the
// round-2 diagnostic (it type-checked); the load itself is emitted as inline asm per
// CDNA5 ISA to avoid builtin-signature guessing.
#if defined(__AMDGCN__) && __has_builtin(__builtin_amdgcn_s_wait_asynccnt)
#define USE_ASYNC 1
#else
#define USE_ASYNC 0
#endif

__device__ __forceinline__ void copy16(h16* lds, const h16* g) {
#if USE_ASYNC
  unsigned l = (unsigned)(size_t)(void*)lds;   // low 32 bits = LDS byte address
  asm volatile("global_load_async_to_lds_b128 %0, %1, off"
               :
               : "v"(l), "v"(g)
               : "memory");
#else
  *(uint4*)lds = *(const uint4*)g;
#endif
}

template <int N>
__device__ __forceinline__ void wait_async() {
#if USE_ASYNC
  __builtin_amdgcn_s_wait_asynccnt(N);   // wait until ASYNCcnt <= N
#endif
}

__device__ __forceinline__ v16h cat8(v8h lo, v8h hi) {
  return __builtin_shufflevector(lo, hi, 0,1,2,3,4,5,6,7,8,9,10,11,12,13,14,15);
}
__device__ __forceinline__ v8f vzero8() {
  v8f z = {0.f,0.f,0.f,0.f,0.f,0.f,0.f,0.f};
  return z;
}
__device__ __forceinline__ v8f wmma_f16(v16h a, v16h b, v8f c) {
  // D = A(16x32 f16) * B(32x16 f16) + C(16x16 f32)
  return __builtin_amdgcn_wmma_f32_16x16x32_f16(false, a, false, b, (short)0, c, false, false);
}

// ---------------- prologue: f32 -> f16 conversions ----------------

__global__ void cvt_kernel(const float* __restrict__ src, h16* __restrict__ dst, int n4) {
  int i = blockIdx.x * blockDim.x + threadIdx.x;
  if (i >= n4) return;
  float4 v = ((const float4*)src)[i];
  v4h o = { (h16)v.x, (h16)v.y, (h16)v.z, (h16)v.w };
  ((v4h*)dst)[i] = o;
}

// hi/lo split (x ~= hi + lo, each f16) for high-accuracy f16 WMMA
__global__ void split_kernel(const float* __restrict__ src,
                             h16* __restrict__ hi, h16* __restrict__ lo, int n) {
  int i = blockIdx.x * blockDim.x + threadIdx.x;
  if (i >= n) return;
  float x = src[i];
  h16 h = (h16)x;
  hi[i] = h;
  lo[i] = (h16)(x - (float)h);
}

// sigma_inv[k,d,e] -> transposed split halves AhT/AlT[k,e,d] (so WMMA B-frags are contiguous in d)
__global__ void splitAt_kernel(const float* __restrict__ A,
                               h16* __restrict__ AhT, h16* __restrict__ AlT) {
  int id = blockIdx.x * blockDim.x + threadIdx.x;   // KC*DIM*DIM threads
  int e = id & (DIM - 1);
  int d = (id >> 7) & (DIM - 1);
  int k = id >> 14;
  float x = A[id];
  h16 h = (h16)x;
  int o = (k * DIM + e) * DIM + d;
  AhT[o] = h;
  AlT[o] = (h16)(x - (float)h);
}

// c1[k,e] = sum_d mu[k,d] * sigma_inv[k,d,e]   (exact f32, tiny)
__global__ void c1_kernel(const float* __restrict__ mu, const float* __restrict__ A,
                          float* __restrict__ c1) {
  int k = blockIdx.x;
  int e = threadIdx.x;
  const float* Ak  = A  + (size_t)k * DIM * DIM;
  const float* muk = mu + (size_t)k * DIM;
  float s = 0.f;
  #pragma unroll 4
  for (int d = 0; d < DIM; ++d) s = fmaf(muk[d], Ak[(size_t)d * DIM + e], s);
  c1[k * DIM + e] = s;
}

// ---------------- psi path: d2[b,k] = || c1[k] - z[b]^T A[k] ||^2 via split-f16 WMMA ----------

__launch_bounds__(256)
__global__ void psi_d2_kernel(const h16* __restrict__ Zh, const h16* __restrict__ Zl,
                              const h16* __restrict__ AhT, const h16* __restrict__ AlT,
                              const float* __restrict__ c1, float* __restrict__ d2) {
  __shared__ __attribute__((aligned(32))) h16 sZh[64 * DIM];
  __shared__ __attribute__((aligned(32))) h16 sZl[64 * DIM];
  __shared__ float sRed[64];
  const int t = threadIdx.x;
  const int k = blockIdx.y;
  const int rowBase = blockIdx.x * 64;
  {
    int row = t >> 2, seg = (t & 3) * 32;
    const uint4* g0 = (const uint4*)(Zh + (size_t)(rowBase + row) * DIM + seg);
    const uint4* g1 = (const uint4*)(Zl + (size_t)(rowBase + row) * DIM + seg);
    uint4* l0 = (uint4*)(sZh + row * DIM + seg);
    uint4* l1 = (uint4*)(sZl + row * DIM + seg);
    #pragma unroll
    for (int j = 0; j < 4; ++j) { l0[j] = g0[j]; l1[j] = g1[j]; }
  }
  if (t < 64) sRed[t] = 0.f;
  __syncthreads();

  const int lane = t & 31;
  const int wave = t >> 5;
  const int r     = wave >> 1;        // subtile row 0..3 (batch)
  const int cbase = (wave & 1) * 4;   // subtile cols (e), 4 per wave
  const int n  = lane & 15;
  const int hi = lane >> 4;

  v8f acc[4];
  #pragma unroll
  for (int c = 0; c < 4; ++c) acc[c] = vzero8();

  #pragma unroll
  for (int ds = 0; ds < 4; ++ds) {                 // K-dim = d, 128 in steps of 32
    const int kk = ds * 32;
    const h16* pa = sZh + (r * 16 + n) * DIM + kk + hi * 16;
    const h16* pl = sZl + (r * 16 + n) * DIM + kk + hi * 16;
    v16h aH = cat8(*(const v8h*)pa, *(const v8h*)(pa + 8));
    v16h aL = cat8(*(const v8h*)pl, *(const v8h*)(pl + 8));
    #pragma unroll
    for (int c = 0; c < 4; ++c) {
      const size_t boff = (size_t)(k * DIM + (cbase + c) * 16 + n) * DIM + kk + hi * 16;
      v16h bH = *(const v16h*)(AhT + boff);
      v16h bL = *(const v16h*)(AlT + boff);
      acc[c] = wmma_f16(aH, bH, acc[c]);
      acc[c] = wmma_f16(aL, bH, acc[c]);
      acc[c] = wmma_f16(aH, bL, acc[c]);
    }
  }

  float sv[8];
  #pragma unroll
  for (int v = 0; v < 8; ++v) sv[v] = 0.f;
  #pragma unroll
  for (int c = 0; c < 4; ++c) {
    float cv = c1[k * DIM + (cbase + c) * 16 + n];   // e depends only on lane N
    #pragma unroll
    for (int v = 0; v < 8; ++v) {
      float df = cv - acc[c][v];
      sv[v] = fmaf(df, df, sv[v]);
    }
  }
  #pragma unroll
  for (int v = 0; v < 8; ++v)
    atomicAdd(&sRed[r * 16 + hi * 8 + v], sv[v]);
  __syncthreads();
  if (t < 64) d2[(size_t)(rowBase + t) * KC + k] = fmaxf(sRed[t], 0.f);
}

__global__ void softmax_kernel(const float* __restrict__ d2, float* __restrict__ psi) {
  __shared__ float s[64];
  int b = blockIdx.x, t = threadIdx.x;   // 64 threads, one k each
  float v = d2[(size_t)b * KC + t];
  s[t] = v; __syncthreads();
  #pragma unroll
  for (int off = 32; off > 0; off >>= 1) {
    if (t < off) s[t] = fminf(s[t], s[t + off]);
    __syncthreads();
  }
  float mn = s[0]; __syncthreads();
  float e = expf(mn - v);                 // softmax(-d2): subtract min(d2)
  s[t] = e; __syncthreads();
  #pragma unroll
  for (int off = 32; off > 0; off >>= 1) {
    if (t < off) s[t] = s[t] + s[t + off];
    __syncthreads();
  }
  psi[(size_t)b * KC + t] = e / s[0];
}

// ---------------- main: out[b,o] = sum_k wgt[b,k] * | sum_i U[b,i] W[k,o,i] | --------------
// 3-deep LDS pipeline over fused (k, i-chunk) steps; async global->LDS staging,
// one barrier per BK=32 step, 8 WMMAs/wave/step, per-k |.|*psi epilogue in registers.

__launch_bounds__(256)
__global__ void moe_abs_gemm_kernel(const h16* __restrict__ U, const h16* __restrict__ W,
                                    const float* __restrict__ wgt, float* __restrict__ out) {
  __shared__ __attribute__((aligned(32))) h16 sU[3][128 * 32];
  __shared__ __attribute__((aligned(32))) h16 sW[3][128 * 32];
  __shared__ float sP[128 * 64];
  const int t = threadIdx.x;
  const int colBase = blockIdx.x * 128;
  const int rowBase = blockIdx.y * 128;
  {
    int row = t >> 1, seg = (t & 1) * 32;
    const float4* g = (const float4*)(wgt + (size_t)(rowBase + row) * KC + seg);
    float4* l = (float4*)(sP + row * KC + seg);
    #pragma unroll
    for (int j = 0; j < 8; ++j) l[j] = g[j];
  }
  const int lane = t & 31;
  const int r  = t >> 5;            // wave = one 16-row subtile strip, all 8 cols
  const int n  = lane & 15;
  const int hi = lane >> 4;
  const int srow = t >> 1;          // staging row
  const int sseg = (t & 1) * 16;    // staging col offset (halfs)

  const h16* gU = U + (size_t)(rowBase + srow) * R1_ + sseg;
  const h16* gW = W + (size_t)(colBase + srow) * R1_ + sseg;
  const int sOff = srow * 32 + sseg;

  // issue one tile's staging (4 x 16B per thread) for pipeline step s into buffer buf
  auto issue = [&](int s, int buf) {
    const int kk = s >> 5;             // cluster index
    const int ii = (s & 31) * 32;      // R1 chunk
    const h16* gu = gU + ii;
    const h16* gw = gW + (size_t)kk * OUT_ * R1_ + ii;
    copy16(&sU[buf][sOff], gu);
    copy16(&sU[buf][sOff + 8], gu + 8);
    copy16(&sW[buf][sOff], gw);
    copy16(&sW[buf][sOff + 8], gw + 8);
  };

  v8f fin[8], acc[8];
  #pragma unroll
  for (int c = 0; c < 8; ++c) { fin[c] = vzero8(); acc[c] = vzero8(); }

  issue(0, 0);
  issue(1, 1);

  int bcur = 0;
  #pragma unroll 1
  for (int step = 0; step < NSTEP; ++step) {
    wait_async<4>();                   // tile `step` landed (leaves step+1 in flight)
    __syncthreads();                   // visible to all waves; prior readers done
    int s2 = step + 2;
    int b2 = bcur + 2; if (b2 >= 3) b2 -= 3;
    if (s2 > NSTEP - 1) s2 = NSTEP - 1;   // harmless duplicate issues at the tail
    issue(s2, b2);

    const h16* bU = sU[bcur];
    const h16* bW = sW[bcur];
    const h16* pa = bU + (r * 16 + n) * 32 + hi * 16;
    v16h a = cat8(*(const v8h*)pa, *(const v8h*)(pa + 8));
    #pragma unroll
    for (int c = 0; c < 8; ++c) {
      v16h b = *(const v16h*)(bW + (c * 16 + n) * 32 + hi * 16);
      acc[c] = wmma_f16(a, b, acc[c]);
    }

    if ((step & 31) == 31) {           // finished full R1 reduction for cluster k
      const int k = step >> 5;
      #pragma unroll
      for (int v = 0; v < 8; ++v) {
        float w = sP[(r * 16 + hi * 8 + v) * KC + k];
        #pragma unroll
        for (int c = 0; c < 8; ++c)
          fin[c][v] = fmaf(w, fabsf(acc[c][v]), fin[c][v]);
      }
      #pragma unroll
      for (int c = 0; c < 8; ++c) acc[c] = vzero8();
    }
    bcur = bcur + 1; if (bcur >= 3) bcur -= 3;
  }

  #pragma unroll
  for (int c = 0; c < 8; ++c) {
    const int ocol = colBase + c * 16 + n;
    #pragma unroll
    for (int v = 0; v < 8; ++v) {
      const int brow = rowBase + r * 16 + hi * 8 + v;
      out[(size_t)brow * OUT_ + ocol] = fin[c][v];
    }
  }
}

// ---------------- host launch ----------------

extern "C" void kernel_launch(void* const* d_in, const int* in_sizes, int n_in,
                              void* d_out, int out_size, void* d_ws, size_t ws_size,
                              hipStream_t stream) {
  (void)in_sizes; (void)n_in; (void)out_size; (void)ws_size;
  const float* z      = (const float*)d_in[0];
  const float* u      = (const float*)d_in[1];
  const float* member = (const float*)d_in[2];
  const float* mu     = (const float*)d_in[3];
  const float* sigma  = (const float*)d_in[4];
  const float* Wcon   = (const float*)d_in[5];
  const float* Wrec   = (const float*)d_in[6];
  float* out = (float*)d_out;
  char* ws = (char*)d_ws;

  const size_t MB = 1ull << 20;
  h16*   Uh  = (h16*)(ws + 0 * MB);                  //   4 MiB
  h16*   Wch = (h16*)(ws + 4 * MB);                  //  64 MiB
  h16*   Wrh = (h16*)(ws + 68 * MB);                 //  64 MiB
  h16*   Zh  = (h16*)(ws + 132 * MB);                // 512 KiB
  h16*   Zl  = Zh + (size_t)B_ * DIM;                // 512 KiB
  h16*   AhT = (h16*)(ws + 133 * MB);                //   2 MiB
  h16*   AlT = (h16*)(ws + 135 * MB);                //   2 MiB
  float* c1  = (float*)(ws + 137 * MB);              //  32 KiB
  float* d2  = (float*)(ws + 137 * MB + (64ull << 10)); // 512 KiB
  float* psi = d2 + (size_t)B_ * KC;                 // 512 KiB  (total ~138.2 MiB)

  const int TPB = 256;
  cvt_kernel<<<(B_ * R1_ / 4) / TPB, TPB, 0, stream>>>(u, Uh, B_ * R1_ / 4);
  cvt_kernel<<<(KC * OUT_ * R1_ / 4) / TPB, TPB, 0, stream>>>(Wcon, Wch, KC * OUT_ * R1_ / 4);
  cvt_kernel<<<(KC * OUT_ * R1_ / 4) / TPB, TPB, 0, stream>>>(Wrec, Wrh, KC * OUT_ * R1_ / 4);
  split_kernel<<<(B_ * DIM) / TPB, TPB, 0, stream>>>(z, Zh, Zl, B_ * DIM);
  splitAt_kernel<<<(KC * DIM * DIM) / TPB, TPB, 0, stream>>>(sigma, AhT, AlT);
  c1_kernel<<<KC, DIM, 0, stream>>>(mu, sigma, c1);
  psi_d2_kernel<<<dim3(B_ / 64, KC), 256, 0, stream>>>(Zh, Zl, AhT, AlT, c1, d2);
  softmax_kernel<<<B_, KC, 0, stream>>>(d2, psi);
  moe_abs_gemm_kernel<<<dim3(OUT_ / 128, B_ / 128), 256, 0, stream>>>(Uh, Wch, psi, out);
  moe_abs_gemm_kernel<<<dim3(OUT_ / 128, B_ / 128), 256, 0, stream>>>(Uh, Wrh, member,
                                                                      out + (size_t)B_ * OUT_);
}
